// BahdanauAttention_83013127897113
// MI455X (gfx1250) — matrix operand order repaired and checked by reading, not compiled, not run on hardware
//
#include <hip/hip_runtime.h>
#include <hip/hip_bf16.h>
#include <math.h>

typedef __attribute__((ext_vector_type(16))) __bf16 v16bf;
typedef __attribute__((ext_vector_type(8)))  __bf16 v8bf;
typedef __attribute__((ext_vector_type(8)))  float  v8f;

constexpr int Bn = 32, Sn = 4096, HIDn = 512, MEMn = 512, ATTn = 512;
constexpr int ROWS_PER_BLOCK = 128;     // 8 waves x 16 rows
constexpr int NT = ATTn / 16;           // 32 N-tiles
constexpr int KS = MEMn / 32;           // 16 K-steps of 32
constexpr int FRAG_BYTES = 1024;        // one 32x16 bf16 B fragment
constexpr int TILE_BYTES = KS * FRAG_BYTES;   // 16 KB per N-tile

#if defined(__HIP_DEVICE_COMPILE__) && __has_builtin(__builtin_amdgcn_tensor_load_to_lds) && __has_builtin(__builtin_amdgcn_s_wait_tensorcnt)
#define USE_TDM 1
#else
#define USE_TDM 0
#endif

// ---------------------------------------------------------------------------
// Kernel 1a: hproj[b][a] = sum_k hidden[b][k] * W_h[k][a]   (tiny GEMM, VALU)
// ---------------------------------------------------------------------------
__global__ __launch_bounds__(256) void hproj_kernel(
    const float* __restrict__ hidden, const float* __restrict__ Wh,
    float* __restrict__ hproj) {
  int idx = blockIdx.x * 256 + threadIdx.x;    // 32*512 = 16384
  int b = idx >> 9, a = idx & 511;
  const float* hrow = hidden + (size_t)b * HIDn;
  float acc = 0.f;
#pragma unroll 4
  for (int k = 0; k < HIDn; ++k)
    acc = fmaf(hrow[k], Wh[(size_t)k * ATTn + a], acc);
  hproj[idx] = acc;
}

// ---------------------------------------------------------------------------
// Kernel 1b: swizzle W_m (K=MEM rows, N=ATT cols, f32) -> bf16 WMMA-B
// fragments.  ISA 7.12.2 B layout (32x16): lane L (half=L>>4, r=L&15) holds
// N = nt*16+r, K = ks*32 + (i>>3)*16 + half*8 + (i&7), i=0..15.
// Fragment storage = [chunk c=i>>3][lane][8 bf16] so LDS reads have 16B lane
// stride (bank-conflict-free ds_load_b128).
// ---------------------------------------------------------------------------
__global__ __launch_bounds__(256) void wswizzle_kernel(
    const float* __restrict__ Wm, __bf16* __restrict__ wt) {
  int flat = blockIdx.x * 256 + threadIdx.x;   // 512*512 = 262144
  int e  = flat & 7;
  int L  = (flat >> 3) & 31;
  int c  = (flat >> 8) & 1;
  int ks = (flat >> 9) & 15;
  int nt = flat >> 13;
  int n = nt * 16 + (L & 15);
  int K = ks * 32 + c * 16 + (L >> 4) * 8 + e;
  wt[flat] = (__bf16)Wm[(size_t)K * ATTn + n];
}

// ---------------------------------------------------------------------------
// TDM: one-shot 1-D descriptor copy of TILE_BYTES from global to LDS.
// Descriptor packing per CDNA5 ISA 8.3/8.4 (group0: count/lds/global/type,
// group1: data_size=8B, tensor_dim0 = tile_dim0 = stride = 2048 units).
// This toolchain's builtin takes 6 args (g0, g1, g2, g3, g4, cpol); the 1-D
// descriptor leaves the trailing groups zero.
// ---------------------------------------------------------------------------
__device__ __forceinline__ void tdm_issue_tile(const void* gsrc, unsigned ldsOff) {
#if USE_TDM
  typedef unsigned int v4u __attribute__((ext_vector_type(4)));
  typedef int v8i __attribute__((ext_vector_type(8)));
  typedef int v4i __attribute__((ext_vector_type(4)));
  const unsigned long long ga = (unsigned long long)gsrc;
  constexpr unsigned UNITS = TILE_BYTES / 8;   // 2048 x 8B
  v4u g0;
  g0[0] = 1u;                                            // count=1, user mode
  g0[1] = ldsOff;                                        // lds_addr (bytes)
  g0[2] = (unsigned)(ga & 0xFFFFFFFFu);                  // global_addr[31:0]
  g0[3] = (unsigned)((ga >> 32) & 0x1FFFFFFu) | (2u << 30); // ga[56:32] | type=2
  v8i g1;
  g1[0] = (3 << 16);                    // workgroup_mask=0, data_size=8B
  g1[1] = (int)(UNITS << 16);           // tensor_dim0[15:0]
  g1[2] = (1 << 16);                    // tensor_dim0 hi=0, tensor_dim1=1
  g1[3] = (int)(UNITS << 16);           // tensor_dim1 hi=0, tile_dim0
  g1[4] = 0;                            // tile_dim1/2 unused
  g1[5] = (int)UNITS;                   // tensor_dim0_stride[31:0]
  g1[6] = 0;
  g1[7] = 0;
  v4i z4 = {0, 0, 0, 0};
  v8i z8 = {0, 0, 0, 0, 0, 0, 0, 0};
  __builtin_amdgcn_tensor_load_to_lds(g0, g1, z4, z4, z8, 0);
#else
  (void)gsrc; (void)ldsOff;
#endif
}

// Build one B fragment from LDS (two conflict-free 16B chunks per lane).
__device__ __forceinline__ v16bf load_bfrag(const unsigned char* fb, int lane) {
  v8bf lo = *(const v8bf*)(fb + lane * 16);
  v8bf hi = *(const v8bf*)(fb + 512 + lane * 16);
  v16bf a;
#pragma unroll
  for (int i = 0; i < 8; ++i) { a[i] = lo[i]; a[i + 8] = hi[i]; }
  return a;
}

// ---------------------------------------------------------------------------
// Kernel 2: fused  score[b][s] = v . tanh( mem[b][s]·W_m + hproj[b] + cov*W_c )
// Each wave holds its 16x512 A slab as 16 resident bf16 fragments (A layout:
// K = ks*32 + half*8 + (i<8 ? i : i+8)).  B tiles are staged into LDS by the
// Tensor Data Mover, double-buffered: issue tile nt+1 while computing nt.
// ---------------------------------------------------------------------------
__global__ __launch_bounds__(256) void score_kernel(
    const float* __restrict__ memory, const float* __restrict__ coverage,
    const float* __restrict__ hproj, const __bf16* __restrict__ wtfrag,
    const float* __restrict__ wc, const float* __restrict__ vvec,
    float* __restrict__ scores) {
  // Single shared blob; TDM descriptors assume the buffers sit at LDS offset 0.
  __shared__ __align__(1024) unsigned char smem[2 * TILE_BYTES + 3 * 2048];
  float* sh_h  = (float*)(smem + 2 * TILE_BYTES);
  float* sh_v  = (float*)(smem + 2 * TILE_BYTES + 2048);
  float* sh_wc = (float*)(smem + 2 * TILE_BYTES + 4096);

  const int tid  = threadIdx.x;
  const int wave = tid >> 5, lane = tid & 31;
  const int half = lane >> 4, r = lane & 15;
  const int blocksPerBatch = Sn / ROWS_PER_BLOCK;        // 32
  const int b  = blockIdx.x / blocksPerBatch;
  const int s0 = (blockIdx.x % blocksPerBatch) * ROWS_PER_BLOCK;
  const int srow = s0 + wave * 16 + r;

  // Kick off the first B tile copy immediately (overlaps A-slab load below).
  const unsigned char* wsrc = (const unsigned char*)wtfrag;
#if USE_TDM
  if (wave == 0) tdm_issue_tile(wsrc, 0u);
#endif

  // --- A slab: 16x512 f32 read once from HBM, kept as bf16 fragments. ---
  const float* arow = memory + ((size_t)b * Sn + srow) * MEMn;
  v16bf afrag[KS];
#pragma unroll
  for (int ks = 0; ks < KS; ++ks) {
    const int d0 = ks * 32 + half * 8;
    v8f lo = *(const v8f*)(arow + d0);          // K = d0..d0+7
    v8f hi = *(const v8f*)(arow + d0 + 16);     // K = d0+16..d0+23
    v16bf a;
#pragma unroll
    for (int i = 0; i < 8; ++i) { a[i] = (__bf16)lo[i]; a[i + 8] = (__bf16)hi[i]; }
    afrag[ks] = a;
  }

  // Stage hproj / v / W_c rows into LDS (read 32x per block otherwise).
  for (int i = tid; i < ATTn; i += 256) {
    sh_h[i]  = hproj[b * ATTn + i];
    sh_v[i]  = vvec[i];
    sh_wc[i] = wc[i];
  }

  float covv[8];
  const float* covRow = coverage + (size_t)b * Sn + s0 + wave * 16 + half * 8;
#pragma unroll
  for (int j = 0; j < 8; ++j) covv[j] = covRow[j];

  float sacc[8];
#pragma unroll
  for (int j = 0; j < 8; ++j) sacc[j] = 0.f;

#if USE_TDM
  if (wave == 0) __builtin_amdgcn_s_wait_tensorcnt(0);   // tile 0 resident
  __syncthreads();
#endif

  for (int nt = 0; nt < NT; ++nt) {
    unsigned char* buf = smem + (nt & 1) * TILE_BYTES;
#if !USE_TDM
    // Fallback: cooperative synchronous copy of this tile.
    {
      const unsigned char* src = wsrc + (size_t)nt * TILE_BYTES;
      for (int o = tid * 32; o < TILE_BYTES; o += 256 * 32)
        *(v16bf*)(buf + o) = *(const v16bf*)(src + o);
      __syncthreads();
    }
#else
    // Prefetch next tile into the other buffer while we compute this one.
    if (wave == 0 && nt + 1 < NT)
      tdm_issue_tile(wsrc + (size_t)(nt + 1) * TILE_BYTES,
                     (unsigned)((nt + 1) & 1) * TILE_BYTES);
#endif

    const int n = nt * 16 + r;                  // C/D layout: N = lane & 15
    const float hv  = sh_h[n];
    const float vv  = sh_v[n];
    const float wcv = sh_wc[n];

    v8f acc = {};
    v16bf bcur = load_bfrag(buf, lane);
#pragma unroll
    for (int ks = 0; ks < KS; ++ks) {
      v16bf bnxt = bcur;
      if (ks + 1 < KS) bnxt = load_bfrag(buf + (ks + 1) * FRAG_BYTES, lane);
      acc = __builtin_amdgcn_wmma_f32_16x16x32_bf16(
          false, afrag[ks], false, bcur, (short)0, acc, false, false);
      bcur = bnxt;
    }

#pragma unroll
    for (int j = 0; j < 8; ++j) {
      float t = tanhf(acc[j] + hv + covv[j] * wcv);
      sacc[j] = fmaf(t, vv, sacc[j]);
    }

#if USE_TDM
    if (wave == 0 && nt + 1 < NT) __builtin_amdgcn_s_wait_tensorcnt(0);
    __syncthreads();                            // next buffer ready / reusable
#else
    __syncthreads();
#endif
  }

  // Reduce over the 16 N-lanes of each half-wave (rows M = half*8 + j).
#pragma unroll
  for (int j = 0; j < 8; ++j) {
    float s = sacc[j];
    s += __shfl_xor(s, 1, 32);
    s += __shfl_xor(s, 2, 32);
    s += __shfl_xor(s, 4, 32);
    s += __shfl_xor(s, 8, 32);
    sacc[j] = s;
  }
  if (r == 0) {
    float* out = scores + (size_t)b * Sn + s0 + wave * 16 + half * 8;
#pragma unroll
    for (int j = 0; j < 8; ++j) out[j] = sacc[j];
  }
}

// ---------------------------------------------------------------------------
// Kernel 3: masked softmax over S per batch; writes weights into d_out.
// ---------------------------------------------------------------------------
__global__ __launch_bounds__(256) void softmax_kernel(
    const float* __restrict__ scores, const unsigned char* __restrict__ pad,
    float* __restrict__ attn) {
  __shared__ float red[256];
  const int b = blockIdx.x, tid = threadIdx.x;
  const float* srow = scores + (size_t)b * Sn;
  const unsigned char* prow = pad + (size_t)b * Sn;
  float lmax = -INFINITY;
  for (int s = tid; s < Sn; s += 256) {
    float v = prow[s] ? -INFINITY : srow[s];
    lmax = fmaxf(lmax, v);
  }
  red[tid] = lmax; __syncthreads();
  for (int off = 128; off > 0; off >>= 1) {
    if (tid < off) red[tid] = fmaxf(red[tid], red[tid + off]);
    __syncthreads();
  }
  const float gmax = red[0]; __syncthreads();

  float* arow = attn + (size_t)b * Sn;
  float lsum = 0.f;
  for (int s = tid; s < Sn; s += 256) {
    float e = prow[s] ? 0.f : __expf(srow[s] - gmax);
    arow[s] = e; lsum += e;
  }
  red[tid] = lsum; __syncthreads();
  for (int off = 128; off > 0; off >>= 1) {
    if (tid < off) red[tid] += red[tid + off];
    __syncthreads();
  }
  const float inv = 1.f / red[0];
  for (int s = tid; s < Sn; s += 256) arow[s] *= inv;
}

// ---------------------------------------------------------------------------
// Kernel 4a/4b: context[b][d] = sum_s attn[b][s]*mem[b][s][d], deterministic
// split over 8 S-chunks (no float atomics -> bit-stable graph replays).
// ---------------------------------------------------------------------------
__global__ __launch_bounds__(128) void context_partial_kernel(
    const float* __restrict__ memory, const float* __restrict__ attn,
    float* __restrict__ partial) {
  const int bid = blockIdx.x;           // 32 * 4 * 8 = 1024
  const int b   = bid >> 5;
  const int dch = (bid >> 3) & 3;
  const int sch = bid & 7;
  const int d = dch * 128 + threadIdx.x;
  const int sbase = sch * (Sn / 8);
  const float* w = attn + (size_t)b * Sn + sbase;
  const float* m = memory + ((size_t)b * Sn + sbase) * MEMn + d;
  float acc = 0.f;
#pragma unroll 4
  for (int s = 0; s < Sn / 8; ++s)
    acc = fmaf(w[s], m[(size_t)s * MEMn], acc);
  partial[((size_t)b * 8 + sch) * MEMn + d] = acc;
}

__global__ __launch_bounds__(256) void context_reduce_kernel(
    const float* __restrict__ partial, float* __restrict__ ctx) {
  int idx = blockIdx.x * 256 + threadIdx.x;    // 16384
  int b = idx >> 9, d = idx & 511;
  float acc = 0.f;
#pragma unroll
  for (int sch = 0; sch < 8; ++sch)
    acc += partial[((size_t)b * 8 + sch) * MEMn + d];
  ctx[idx] = acc;
}

// ---------------------------------------------------------------------------
extern "C" void kernel_launch(void* const* d_in, const int* in_sizes, int n_in,
                              void* d_out, int out_size, void* d_ws, size_t ws_size,
                              hipStream_t stream) {
  const float* hidden   = (const float*)d_in[0];
  const float* memory   = (const float*)d_in[1];
  const unsigned char* mem_pad = (const unsigned char*)d_in[2];
  const float* coverage = (const float*)d_in[3];
  const float* Wh       = (const float*)d_in[4];
  const float* Wm       = (const float*)d_in[5];
  const float* Wc       = (const float*)d_in[6];
  const float* vvec     = (const float*)d_in[7];

  char* ws = (char*)d_ws;
  float*  hproj   = (float*)(ws);                        // 64 KB
  __bf16* wtfrag  = (__bf16*)(ws + 65536);               // 512 KB
  float*  scores  = (float*)(ws + 65536 + 524288);       // 512 KB
  float*  partial = (float*)(ws + 65536 + 2 * 524288);   // 512 KB

  float* ctx  = (float*)d_out;                 // 32*512
  float* attn = (float*)d_out + Bn * MEMn;     // 32*4096

  hproj_kernel<<<64, 256, 0, stream>>>(hidden, Wh, hproj);
  wswizzle_kernel<<<1024, 256, 0, stream>>>(Wm, wtfrag);
  score_kernel<<<Bn * (Sn / ROWS_PER_BLOCK), 256, 0, stream>>>(
      memory, coverage, hproj, wtfrag, Wc, vvec, scores);
  softmax_kernel<<<Bn, 256, 0, stream>>>(scores, mem_pad, attn);
  context_partial_kernel<<<1024, 128, 0, stream>>>(memory, attn, partial);
  context_reduce_kernel<<<64, 256, 0, stream>>>(partial, ctx);
}
